// JetAssignmentTransformer_44813688767268
// MI455X (gfx1250) — matrix operand order repaired and checked by reading, not compiled, not run on hardware
//
#include <hip/hip_runtime.h>
#include <math.h>

// ---------------------------------------------------------------------------
// Types for CDNA5 WMMA (wave32): D(f32 16x16) = A(bf16 16x32) x B(bf16 32x16) + C
// ---------------------------------------------------------------------------
typedef __attribute__((ext_vector_type(16))) __bf16       v16bf;
typedef __attribute__((ext_vector_type(8)))  float        v8f;
typedef __attribute__((ext_vector_type(8)))  unsigned int v8u;

#define NB    2048   // batch
#define NJ    7      // jets
#define ND    128    // model dim
#define NH    8      // encoder heads
#define NL    4      // encoder layers
#define NDFF  512
#define NTRIP 35
#define NASGN 70
#define KPAD_S1 672  // 664 padded to 21*32

__device__ __forceinline__ unsigned short f2bf(float f) {
  unsigned int u = __builtin_bit_cast(unsigned int, f);
  unsigned int r = u + 0x7fffu + ((u >> 16) & 1u);
  return (unsigned short)(r >> 16);
}
__device__ __forceinline__ float bf2f(unsigned short h) {
  unsigned int u = ((unsigned int)h) << 16;
  return __builtin_bit_cast(float, u);
}
__device__ __forceinline__ float gelu_f(float x) {
  return 0.5f * x * (1.f + erff(x * 0.70710678118654752f));
}

// A fragment: 16x32 bf16, row-major source with leading dim lda (elements).
// ISA layout: lanes 0-15 -> M=lane, Kbase=0 ; lanes 16-31 -> M=lane-16, Kbase=8
//   VGPR v<4: K = Kbase+2v ; v>=4: K = 16+Kbase+2(v-4); each VGPR packs (K,K+1).
__device__ __forceinline__ v16bf load_a_frag(const unsigned short* A, int lda,
                                             int kk, int lane) {
  int m  = lane & 15;
  int kb = (lane < 16) ? 0 : 8;
  const unsigned short* row = A + (size_t)m * lda + kk;
  v8u u;
#pragma unroll
  for (int v = 0; v < 8; ++v) {
    int k = (v < 4) ? (kb + 2 * v) : (16 + kb + 2 * (v - 4));
    u[v] = *(const unsigned int*)(row + k);
  }
  return __builtin_bit_cast(v16bf, u);
}
// B fragment: 32x16 bf16. Weights pre-transposed to [N,K] so lane n = lane&15,
// same K layout as A: Kbase=0/8 per half, VGPR v<4: K=Kbase+2v, v>=4: K=16+Kbase+2(v-4).
__device__ __forceinline__ v16bf load_b_frag(const unsigned short* Bt, int ldb,
                                             int kk, int lane) {
  int n  = lane & 15;
  int kb = (lane < 16) ? 0 : 8;
  const unsigned short* row = Bt + (size_t)n * ldb + kk + kb;
  v8u u;
#pragma unroll
  for (int v = 0; v < 8; ++v) u[v] = *(const unsigned int*)(row + 2 * v + ((v < 4) ? 0 : 8));
  return __builtin_bit_cast(v16bf, u);
}
__device__ __forceinline__ v8f wmma_bf16(v16bf a, v16bf b, v8f c) {
  return __builtin_amdgcn_wmma_f32_16x16x32_bf16(false, a, false, b, (short)0,
                                                 c, false, false);
}

// ---------------------------------------------------------------------------
// Generic row-batched GEMM: out = epilogue(A[M,K]bf16 @ Wt[N,K]bf16^T + bias)
// grid = (N/64, M/16), block = 32: one wave computes a 16x64 strip (4 tiles),
// loading the A fragment once per K-step and amortizing it over 4 WMMAs.
// MODE 0: f32 store. 1: bf16 store. 2: relu->bf16. 3: gelu->bf16. 4: f32 +=
// ---------------------------------------------------------------------------
template <int MODE>
__global__ __launch_bounds__(32)
void gemm_bf16_kernel(const unsigned short* __restrict__ A,
                      const unsigned short* __restrict__ Bt,
                      const float* __restrict__ bias, void* __restrict__ Out,
                      int M, int N, int K) {
  int nb = blockIdx.x, mt = blockIdx.y;
  int lane = threadIdx.x;
  const unsigned short* Ab = A + (size_t)mt * 16 * K;
  const unsigned short* Bb = Bt + (size_t)nb * 64 * K;
  v8f acc[4] = {{}, {}, {}, {}};
  for (int kk = 0; kk < K; kk += 32) {
    v16bf a = load_a_frag(Ab, K, kk, lane);
#pragma unroll
    for (int i = 0; i < 4; ++i)
      acc[i] = wmma_bf16(a, load_b_frag(Bb + (size_t)i * 16 * K, K, kk, lane),
                         acc[i]);
  }
  int m0 = mt * 16 + ((lane < 16) ? 0 : 8);
  int nn = lane & 15;
#pragma unroll
  for (int i = 0; i < 4; ++i) {
    int n = (nb * 4 + i) * 16 + nn;
    float bv = bias[n];
#pragma unroll
    for (int r = 0; r < 8; ++r) {
      float val  = acc[i][r] + bv;
      size_t idx = (size_t)(m0 + r) * N + n;
      if (MODE == 0) ((float*)Out)[idx] = val;
      else if (MODE == 1) ((unsigned short*)Out)[idx] = f2bf(val);
      else if (MODE == 2) ((unsigned short*)Out)[idx] = f2bf(fmaxf(val, 0.f));
      else if (MODE == 3) ((unsigned short*)Out)[idx] = f2bf(gelu_f(val));
      else ((float*)Out)[idx] += val;
    }
  }
}

// ---------------------------------------------------------------------------
// Combinatorial tables (deterministic, rebuilt every call)
// ---------------------------------------------------------------------------
__device__ int trip_id_of(const int* trips, int a, int b, int c) {
  for (int t = 0; t < NTRIP; ++t)
    if (trips[t * 3] == a && trips[t * 3 + 1] == b && trips[t * 3 + 2] == c)
      return t;
  return 0;
}
__global__ void build_tables_kernel(int* trips, int* isr, int* g1t, int* g2t,
                                    int* g1j, int* g2j) {
  if (threadIdx.x != 0 || blockIdx.x != 0) return;
  int nt = 0;
  for (int a = 0; a < NJ; ++a)
    for (int b = a + 1; b < NJ; ++b)
      for (int c = b + 1; c < NJ; ++c) {
        trips[nt * 3] = a; trips[nt * 3 + 1] = b; trips[nt * 3 + 2] = c; ++nt;
      }
  int na = 0;
  for (int is = 0; is < NJ; ++is) {
    int rem[6], k = 0;
    for (int j = 0; j < NJ; ++j) if (j != is) rem[k++] = j;
    for (int i1 = 0; i1 < 6; ++i1)
      for (int i2 = i1 + 1; i2 < 6; ++i2)
        for (int i3 = i2 + 1; i3 < 6; ++i3) {
          int c0 = rem[i1], c1 = rem[i2], c2 = rem[i3];
          int comp[3], m = 0;
          for (int q = 0; q < 6; ++q) {
            int jv = rem[q];
            if (jv != c0 && jv != c1 && jv != c2) comp[m++] = jv;
          }
          bool lt = (c0 < comp[0]) ||
                    (c0 == comp[0] && (c1 < comp[1] ||
                                       (c1 == comp[1] && c2 < comp[2])));
          if (!lt) continue;
          isr[na] = is;
          g1j[na * 3] = c0; g1j[na * 3 + 1] = c1; g1j[na * 3 + 2] = c2;
          g2j[na * 3] = comp[0]; g2j[na * 3 + 1] = comp[1]; g2j[na * 3 + 2] = comp[2];
          g1t[na] = trip_id_of(trips, c0, c1, c2);
          g2t[na] = trip_id_of(trips, comp[0], comp[1], comp[2]);
          ++na;
        }
  }
}

// Weight prep: out[n*Kpad + k] = bf16(W[k*N + n]), zero-padded K.
__global__ void wprep_kernel(const float* __restrict__ W,
                             unsigned short* __restrict__ out, int K, int N,
                             int Kpad) {
  int idx = blockIdx.x * 256 + threadIdx.x;
  if (idx >= N * Kpad) return;
  int n = idx / Kpad, k = idx - n * Kpad;
  float v = (k < K) ? W[(size_t)k * N + n] : 0.f;
  out[idx] = f2bf(v);
}

// ---------------------------------------------------------------------------
// Featurize + attention bias (one block per event)
// ---------------------------------------------------------------------------
__global__ __launch_bounds__(128)
void featurize_kernel(const float* __restrict__ fm, const float* __restrict__ W_in,
                      const float* __restrict__ b_in, const float* __restrict__ pt_w,
                      const float* __restrict__ dr_w, const float* __restrict__ mass_w,
                      float* __restrict__ x, float* __restrict__ biasb) {
  __shared__ float jf[NJ][8];   // E px py pz log_pt eta phi m_jet
  __shared__ float aux[NJ][2];  // cphi sphi
  int b = blockIdx.x, t = threadIdx.x;
  if (t < NJ) {
    const float* p = fm + ((size_t)b * NJ + t) * 4;
    float E = p[0], px = p[1], py = p[2], pz = p[3];
    float pt = fmaxf(sqrtf(px * px + py * py), 1e-8f);
    jf[t][0] = E; jf[t][1] = px; jf[t][2] = py; jf[t][3] = pz;
    jf[t][4] = logf(pt);
    jf[t][5] = asinhf(pz / pt);
    jf[t][6] = atan2f(py, px);
    jf[t][7] = sqrtf(fmaxf(E * E - px * px - py * py - pz * pz, 1e-8f));
    aux[t][0] = px / pt; aux[t][1] = py / pt;
  }
  __syncthreads();
  for (int j = 0; j < NJ; ++j) {
    float acc = b_in[t];
#pragma unroll
    for (int i = 0; i < 8; ++i) acc += jf[j][i] * W_in[i * ND + t];
    x[((size_t)b * NJ + j) * ND + t] = acc;
  }
  if (t < 49) {
    int i = t / 7, j = t - i * 7;
    float ptb = jf[i][4] - jf[j][4];
    float de  = jf[i][5] - jf[j][5];
    float cd  = aux[i][0] * aux[j][0] + aux[i][1] * aux[j][1];
    cd = fminf(fmaxf(cd, -1.f + 1e-7f), 1.f - 1e-7f);
    float ac = acosf(cd);
    float dr = sqrtf(de * de + ac * ac);
    float Es = jf[i][0] + jf[j][0];
    float sx = jf[i][1] + jf[j][1], sy = jf[i][2] + jf[j][2], sz = jf[i][3] + jf[j][3];
    float mb = 0.5f * logf(fmaxf(Es * Es - sx * sx - sy * sy - sz * sz, 1e-8f));
    for (int h = 0; h < NH; ++h)
      biasb[(((size_t)b * NH + h) * 7 + i) * 7 + j] =
          pt_w[h] * ptb + dr_w[h] * dr + mass_w[h] * mb;
  }
}

// LayerNorm over D=128 rows -> bf16 (and optionally f32)
template <bool WF32>
__global__ __launch_bounds__(128)
void ln_kernel(const float* __restrict__ X, const float* __restrict__ g,
               const float* __restrict__ bt, float* __restrict__ Yf,
               unsigned short* __restrict__ Yb) {
  int row = blockIdx.x, c = threadIdx.x;
  float v = X[(size_t)row * ND + c];
  float s = v, ss = v * v;
  for (int o = 16; o; o >>= 1) { s += __shfl_xor(s, o, 32); ss += __shfl_xor(ss, o, 32); }
  __shared__ float red[8];
  int w = c >> 5, lane = c & 31;
  if (lane == 0) { red[w] = s; red[4 + w] = ss; }
  __syncthreads();
  s  = red[0] + red[1] + red[2] + red[3];
  ss = red[4] + red[5] + red[6] + red[7];
  float mu  = s * (1.f / ND);
  float inv = rsqrtf(ss * (1.f / ND) - mu * mu + 1e-5f);
  float y = (v - mu) * inv * g[c] + bt[c];
  if (WF32) Yf[(size_t)row * ND + c] = y;
  Yb[(size_t)row * ND + c] = f2bf(y);
}

// Encoder self-attention (7 tokens, 8 heads, dh=16) with pairwise bias.
__global__ __launch_bounds__(64)
void enc_attn_kernel(const float* __restrict__ qkv, const float* __restrict__ biasb,
                     unsigned short* __restrict__ attn) {
  int b = blockIdx.x, t = threadIdx.x;
  if (t >= NH * NJ) return;
  int h = t / 7, q = t - h * 7;
  const float* qv = qkv + ((size_t)b * NJ + q) * 384 + h * 16;
  float s[7], mx = -1e30f;
  for (int j = 0; j < NJ; ++j) {
    const float* kv = qkv + ((size_t)b * NJ + j) * 384 + 128 + h * 16;
    float d = 0.f;
#pragma unroll
    for (int e = 0; e < 16; ++e) d += qv[e] * kv[e];
    s[j] = d * 0.25f + biasb[(((size_t)b * NH + h) * 7 + q) * 7 + j];
    mx = fmaxf(mx, s[j]);
  }
  float den = 0.f;
  for (int j = 0; j < NJ; ++j) { s[j] = __expf(s[j] - mx); den += s[j]; }
  float inv = 1.f / den;
  float o[16] = {};
  for (int j = 0; j < NJ; ++j) {
    const float* vv = qkv + ((size_t)b * NJ + j) * 384 + 256 + h * 16;
#pragma unroll
    for (int e = 0; e < 16; ++e) o[e] += s[j] * vv[e];
  }
#pragma unroll
  for (int e = 0; e < 16; ++e)
    attn[((size_t)b * NJ + q) * ND + h * 16 + e] = f2bf(o[e] * inv);
}

// triplet_emb = mean of 3 jet embeddings -> bf16
__global__ void triplet_mean_kernel(const float* __restrict__ x,
                                    const int* __restrict__ trips,
                                    unsigned short* __restrict__ out) {
  int idx = blockIdx.x * 256 + threadIdx.x;
  if (idx >= NB * NTRIP * ND) return;
  int c = idx & 127, gs = idx >> 7;
  int b = gs / NTRIP, tt = gs - b * NTRIP;
  const float* xb = x + (size_t)b * NJ * ND + c;
  float v = (xb[trips[tt * 3] * ND] + xb[trips[tt * 3 + 1] * ND] +
             xb[trips[tt * 3 + 2] * ND]) * (1.f / 3.f);
  out[idx] = f2bf(v);
}

__global__ void f32_to_bf16_kernel(const float* __restrict__ in,
                                   unsigned short* __restrict__ out, int n) {
  int idx = blockIdx.x * 256 + threadIdx.x;
  if (idx < n) out[idx] = f2bf(in[idx]);
}

// Triplet cross-attention: jets attend to the 15 triplets containing them.
__global__ __launch_bounds__(32)
void tca_attn_kernel(const float* __restrict__ q, const unsigned short* __restrict__ kb,
                     const unsigned short* __restrict__ vb, const int* __restrict__ trips,
                     unsigned short* __restrict__ attn) {
  int b = blockIdx.x, t = threadIdx.x;
  if (t >= 4 * NJ) return;
  int h = t / 7, j = t - h * 7;
  const float* qv = q + ((size_t)b * NJ + j) * ND + h * 32;
  float sc[NTRIP], mx = -1e30f;
  for (int tt = 0; tt < NTRIP; ++tt) {
    bool mem = (trips[tt * 3] == j) | (trips[tt * 3 + 1] == j) | (trips[tt * 3 + 2] == j);
    if (!mem) { sc[tt] = -1e30f; continue; }
    const unsigned short* kv = kb + ((size_t)b * NTRIP + tt) * ND + h * 32;
    float d = 0.f;
#pragma unroll
    for (int e = 0; e < 32; ++e) d += qv[e] * bf2f(kv[e]);
    sc[tt] = d * 0.17677669529663687f;
    mx = fmaxf(mx, sc[tt]);
  }
  float den = 0.f;
  for (int tt = 0; tt < NTRIP; ++tt) {
    float e = (sc[tt] > -1e29f) ? __expf(sc[tt] - mx) : 0.f;
    sc[tt] = e; den += e;
  }
  float inv = 1.f / den;
  float o[32] = {};
  for (int tt = 0; tt < NTRIP; ++tt) {
    float wgt = sc[tt];
    if (wgt == 0.f) continue;
    const unsigned short* vv = vb + ((size_t)b * NTRIP + tt) * ND + h * 32;
#pragma unroll
    for (int e = 0; e < 32; ++e) o[e] += wgt * bf2f(vv[e]);
  }
#pragma unroll
  for (int e = 0; e < 32; ++e)
    attn[((size_t)b * NJ + j) * ND + h * 32 + e] = f2bf(o[e] * inv);
}

// ---------------------------------------------------------------------------
// Fused group transformer: 16 triplet-sequences (48 tokens) per workgroup,
// whole encoder layer in LDS, weights streamed from L2 as WMMA B operands.
// ---------------------------------------------------------------------------
__device__ __forceinline__ void ln48_lds(const float* xs, unsigned short* hb,
                                         const float* g, const float* bt,
                                         int lane, int w) {
  for (int r = w; r < 48; r += 4) {
    const float* row = xs + r * ND;
    float v[4], s = 0.f, ss = 0.f;
#pragma unroll
    for (int q = 0; q < 4; ++q) {
      v[q] = row[lane + 32 * q]; s += v[q]; ss += v[q] * v[q];
    }
    for (int o = 16; o; o >>= 1) { s += __shfl_xor(s, o, 32); ss += __shfl_xor(ss, o, 32); }
    float mu  = s * (1.f / ND);
    float inv = rsqrtf(ss * (1.f / ND) - mu * mu + 1e-5f);
#pragma unroll
    for (int q = 0; q < 4; ++q) {
      int c = lane + 32 * q;
      hb[r * ND + c] = f2bf((v[q] - mu) * inv * g[c] + bt[c]);
    }
  }
}

__global__ __launch_bounds__(128)
void gt_fused_kernel(const float* __restrict__ z,
                     const unsigned short* __restrict__ Wqkv_t, const float* __restrict__ bqkv,
                     const unsigned short* __restrict__ Wo_t, const float* __restrict__ bo,
                     const unsigned short* __restrict__ W1_t, const float* __restrict__ bb1,
                     const unsigned short* __restrict__ W2_t, const float* __restrict__ bb2,
                     const float* __restrict__ g1, const float* __restrict__ b1,
                     const float* __restrict__ g2, const float* __restrict__ b2,
                     const int* __restrict__ trips, float* __restrict__ gemb) {
  __shared__ __align__(16) unsigned char sraw[48 * ND * 4 + 48 * ND * 2 +
                                              48 * 384 * 4 + 48 * ND * 2];
  float* xs           = (float*)sraw;                                   // 48x128 f32
  unsigned short* hb  = (unsigned short*)(sraw + 48 * ND * 4);          // 48x128 bf16
  float* qkvs         = (float*)(sraw + 48 * ND * 4 + 48 * ND * 2);     // 48x384 f32
  unsigned short* ffs = (unsigned short*)qkvs;                          // alias 48x256 bf16
  unsigned short* ats = (unsigned short*)(sraw + 48 * ND * 4 + 48 * ND * 2 + 48 * 384 * 4);
  int tid = threadIdx.x, lane = tid & 31, w = tid >> 5;
  int seq0 = blockIdx.x * 16;
  // gather token rows from jet_emb
  for (int idx = tid; idx < 48 * ND; idx += 128) {
    int r = idx >> 7, c = idx & 127;
    int gseq = seq0 + r / 3, tok = r - (r / 3) * 3;
    int b = gseq / NTRIP, tt = gseq - b * NTRIP;
    xs[idx] = z[((size_t)b * NJ + trips[tt * 3 + tok]) * ND + c];
  }
  __syncthreads();
  ln48_lds(xs, hb, g1, b1, lane, w);
  __syncthreads();
  // QKV: 3 mtiles x 24 ntiles
  for (int ti = w; ti < 72; ti += 4) {
    int mt = ti / 24, nt = ti - mt * 24;
    v8f acc = {};
    const unsigned short* Ab = hb + mt * 16 * ND;
    const unsigned short* Bb = Wqkv_t + (size_t)nt * 16 * ND;
    for (int kk = 0; kk < ND; kk += 32)
      acc = wmma_bf16(load_a_frag(Ab, ND, kk, lane), load_b_frag(Bb, ND, kk, lane), acc);
    int n = nt * 16 + (lane & 15);
    int m0 = mt * 16 + ((lane < 16) ? 0 : 8);
    float bv = bqkv[n];
#pragma unroll
    for (int r = 0; r < 8; ++r) qkvs[(m0 + r) * 384 + n] = acc[r] + bv;
  }
  __syncthreads();
  // attention: 3x3 per (sequence, head), dh=32
  if (tid < 64) {
    int si = tid >> 2, h = tid & 3;
    int base = si * 3;
    float sc[3][3];
    for (int i = 0; i < 3; ++i)
      for (int j = 0; j < 3; ++j) {
        float d = 0.f;
#pragma unroll
        for (int e = 0; e < 32; ++e)
          d += qkvs[(base + i) * 384 + h * 32 + e] *
               qkvs[(base + j) * 384 + 128 + h * 32 + e];
        sc[i][j] = d * 0.17677669529663687f;
      }
    for (int i = 0; i < 3; ++i) {
      float mx = fmaxf(sc[i][0], fmaxf(sc[i][1], sc[i][2]));
      float e0 = __expf(sc[i][0] - mx), e1 = __expf(sc[i][1] - mx), e2 = __expf(sc[i][2] - mx);
      float inv = 1.f / (e0 + e1 + e2);
#pragma unroll
      for (int e = 0; e < 32; ++e) {
        float o = e0 * qkvs[(base + 0) * 384 + 256 + h * 32 + e] +
                  e1 * qkvs[(base + 1) * 384 + 256 + h * 32 + e] +
                  e2 * qkvs[(base + 2) * 384 + 256 + h * 32 + e];
        ats[(base + i) * ND + h * 32 + e] = f2bf(o * inv);
      }
    }
  }
  __syncthreads();
  // Wo + residual into xs
  for (int ti = w; ti < 24; ti += 4) {
    int mt = ti / 8, nt = ti - mt * 8;
    v8f acc = {};
    const unsigned short* Ab = ats + mt * 16 * ND;
    const unsigned short* Bb = Wo_t + (size_t)nt * 16 * ND;
    for (int kk = 0; kk < ND; kk += 32)
      acc = wmma_bf16(load_a_frag(Ab, ND, kk, lane), load_b_frag(Bb, ND, kk, lane), acc);
    int n = nt * 16 + (lane & 15);
    int m0 = mt * 16 + ((lane < 16) ? 0 : 8);
    float bv = bo[n];
#pragma unroll
    for (int r = 0; r < 8; ++r) xs[(m0 + r) * ND + n] += acc[r] + bv;
  }
  __syncthreads();
  ln48_lds(xs, hb, g2, b2, lane, w);
  __syncthreads();
  // FFN1: N=256, relu -> bf16 (ffs aliases qkvs; qkv dead by now)
  for (int ti = w; ti < 48; ti += 4) {
    int mt = ti / 16, nt = ti - mt * 16;
    v8f acc = {};
    const unsigned short* Ab = hb + mt * 16 * ND;
    const unsigned short* Bb = W1_t + (size_t)nt * 16 * ND;
    for (int kk = 0; kk < ND; kk += 32)
      acc = wmma_bf16(load_a_frag(Ab, ND, kk, lane), load_b_frag(Bb, ND, kk, lane), acc);
    int n = nt * 16 + (lane & 15);
    int m0 = mt * 16 + ((lane < 16) ? 0 : 8);
    float bv = bb1[n];
#pragma unroll
    for (int r = 0; r < 8; ++r)
      ffs[(m0 + r) * 256 + n] = f2bf(fmaxf(acc[r] + bv, 0.f));
  }
  __syncthreads();
  // FFN2: K=256, residual into xs
  for (int ti = w; ti < 24; ti += 4) {
    int mt = ti / 8, nt = ti - mt * 8;
    v8f acc = {};
    const unsigned short* Ab = ffs + mt * 16 * 256;
    const unsigned short* Bb = W2_t + (size_t)nt * 16 * 256;
    for (int kk = 0; kk < 256; kk += 32)
      acc = wmma_bf16(load_a_frag(Ab, 256, kk, lane), load_b_frag(Bb, 256, kk, lane), acc);
    int n = nt * 16 + (lane & 15);
    int m0 = mt * 16 + ((lane < 16) ? 0 : 8);
    float bv = bb2[n];
#pragma unroll
    for (int r = 0; r < 8; ++r) xs[(m0 + r) * ND + n] += acc[r] + bv;
  }
  __syncthreads();
  // mean over 3 tokens
  for (int idx = tid; idx < 16 * ND; idx += 128) {
    int si = idx >> 7, c = idx & 127;
    float m = (xs[(si * 3 + 0) * ND + c] + xs[(si * 3 + 1) * ND + c] +
               xs[(si * 3 + 2) * ND + c]) * (1.f / 3.f);
    gemb[(size_t)(seq0 + si) * ND + c] = m;
  }
}

__global__ __launch_bounds__(128)
void event_mean_kernel(const float* __restrict__ z, float* __restrict__ ev) {
  int b = blockIdx.x, c = threadIdx.x;
  float s = 0.f;
  for (int j = 0; j < NJ; ++j) s += z[((size_t)b * NJ + j) * ND + c];
  ev[(size_t)b * ND + c] = s * (1.f / NJ);
}

__device__ __forceinline__ void feat6_dev(const float* p, float* o) {
  float E = p[0], px = p[1], py = p[2], pz = p[3];
  float pt = sqrtf(px * px + py * py);
  float p3 = sqrtf(pt * pt + pz * pz);
  o[0] = sqrtf(fmaxf(E * E - p3 * p3, 1e-8f));
  o[1] = pt;
  o[2] = asinhf(pz / fmaxf(pt, 1e-8f));
  o[3] = atan2f(py, px);
  o[4] = E;
  o[5] = p3;
}
__global__ void physics_kernel(const float* __restrict__ fm, const int* __restrict__ isr,
                               const int* __restrict__ g1j, const int* __restrict__ g2j,
                               const float* __restrict__ ph_g, const float* __restrict__ ph_b,
                               float* __restrict__ phys) {
  int idx = blockIdx.x * 256 + threadIdx.x;
  if (idx >= NB * NASGN) return;
  int b = idx / NASGN, a = idx - b * NASGN;
  const float* f = fm + (size_t)b * NJ * 4;
  float p1[4] = {}, p2[4] = {}, pi[4], ps[4];
  for (int q = 0; q < 3; ++q) {
    int j1 = g1j[a * 3 + q], j2 = g2j[a * 3 + q];
#pragma unroll
    for (int e = 0; e < 4; ++e) { p1[e] += f[j1 * 4 + e]; p2[e] += f[j2 * 4 + e]; }
  }
  int ji = isr[a];
#pragma unroll
  for (int e = 0; e < 4; ++e) { pi[e] = f[ji * 4 + e]; ps[e] = p1[e] + p2[e]; }
  float ft[24];
  feat6_dev(p1, ft + 0); feat6_dev(p2, ft + 6);
  feat6_dev(pi, ft + 12); feat6_dev(ps, ft + 18);
  float s = 0.f;
  for (int e = 0; e < 24; ++e) s += ft[e];
  float mu = s * (1.f / 24.f), ss = 0.f;
  for (int e = 0; e < 24; ++e) { float d = ft[e] - mu; ss += d * d; }
  float inv = rsqrtf(ss * (1.f / 24.f) + 1e-5f);
  for (int e = 0; e < 24; ++e)
    phys[(size_t)idx * 24 + e] = (ft[e] - mu) * inv * ph_g[e] + ph_b[e];
}

// ---------------------------------------------------------------------------
// Fused scoring head: build 16x672 feats tile in LDS, WMMA 672->256->128, dot.
// ---------------------------------------------------------------------------
__global__ __launch_bounds__(128)
void score_fused_kernel(const float* __restrict__ z, const float* __restrict__ gemb,
                        const float* __restrict__ evm, const float* __restrict__ phys,
                        const unsigned short* __restrict__ W1t, const float* __restrict__ b1,
                        const unsigned short* __restrict__ W2t, const float* __restrict__ b2,
                        const float* __restrict__ W3, const float* __restrict__ b3,
                        const int* __restrict__ isr, const int* __restrict__ g1t,
                        const int* __restrict__ g2t, float* __restrict__ out) {
  __shared__ __align__(16) unsigned short As[16 * KPAD_S1];
  __shared__ __align__(16) unsigned short h1s[16 * 256];
  __shared__ __align__(16) float h2s[16 * ND];
  int tid = threadIdx.x, lane = tid & 31, w = tid >> 5;
  int row0 = blockIdx.x * 16;
  for (int idx = tid; idx < 16 * KPAD_S1; idx += 128) {
    int r = idx / KPAD_S1, c = idx - r * KPAD_S1;
    int gr = row0 + r, b = gr / NASGN, a = gr - b * NASGN;
    float v;
    if (c < 128) {
      v = z[((size_t)b * NJ + isr[a]) * ND + c];
    } else if (c < 512) {
      int cc = c & 127, sel = c >> 7;  // 1=sum 2=prod 3=sqdiff
      float ga = gemb[((size_t)b * NTRIP + g1t[a]) * ND + cc];
      float gb = gemb[((size_t)b * NTRIP + g2t[a]) * ND + cc];
      v = (sel == 1) ? (ga + gb) : (sel == 2) ? (ga * gb) : (ga - gb) * (ga - gb);
    } else if (c < 640) {
      v = evm[(size_t)b * ND + (c - 512)];
    } else if (c < 664) {
      v = phys[((size_t)b * NASGN + a) * 24 + (c - 640)];
    } else {
      v = 0.f;
    }
    As[idx] = f2bf(v);
  }
  __syncthreads();
  {  // h1 = gelu(A @ s_W1 + b1): 16 ntiles, 4 per wave, K=672
    v8f acc[4] = {{}, {}, {}, {}};
    for (int kk = 0; kk < KPAD_S1; kk += 32) {
      v16bf a = load_a_frag(As, KPAD_S1, kk, lane);
#pragma unroll
      for (int i = 0; i < 4; ++i) {
        int nt = w * 4 + i;
        acc[i] = wmma_bf16(a, load_b_frag(W1t + (size_t)nt * 16 * KPAD_S1, KPAD_S1, kk, lane), acc[i]);
      }
    }
    int m0 = (lane < 16) ? 0 : 8, nn = lane & 15;
#pragma unroll
    for (int i = 0; i < 4; ++i) {
      int n = (w * 4 + i) * 16 + nn;
      float bv = b1[n];
#pragma unroll
      for (int r = 0; r < 8; ++r)
        h1s[(m0 + r) * 256 + n] = f2bf(gelu_f(acc[i][r] + bv));
    }
  }
  __syncthreads();
  {  // h2 = gelu(h1 @ s_W2 + b2): 8 ntiles, 2 per wave, K=256
    v8f acc[2] = {{}, {}};
    for (int kk = 0; kk < 256; kk += 32) {
      v16bf a = load_a_frag(h1s, 256, kk, lane);
#pragma unroll
      for (int i = 0; i < 2; ++i) {
        int nt = w * 2 + i;
        acc[i] = wmma_bf16(a, load_b_frag(W2t + (size_t)nt * 16 * 256, 256, kk, lane), acc[i]);
      }
    }
    int m0 = (lane < 16) ? 0 : 8, nn = lane & 15;
#pragma unroll
    for (int i = 0; i < 2; ++i) {
      int n = (w * 2 + i) * 16 + nn;
      float bv = b2[n];
#pragma unroll
      for (int r = 0; r < 8; ++r)
        h2s[(m0 + r) * ND + n] = gelu_f(acc[i][r] + bv);
    }
  }
  __syncthreads();
  if (tid < 16) {
    float d = b3[0];
    for (int c = 0; c < ND; ++c) d += h2s[tid * ND + c] * W3[c];
    out[row0 + tid] = d;
  }
}

// ---------------------------------------------------------------------------
// Host orchestration
// ---------------------------------------------------------------------------
extern "C" void kernel_launch(void* const* d_in, const int* in_sizes, int n_in,
                              void* d_out, int out_size, void* d_ws, size_t ws_size,
                              hipStream_t stream) {
  (void)in_sizes; (void)n_in; (void)out_size; (void)ws_size;
  const float* fm       = (const float*)d_in[0];
  const float* W_in     = (const float*)d_in[1];
  const float* b_in     = (const float*)d_in[2];
  const float* pt_w     = (const float*)d_in[3];
  const float* dr_w     = (const float*)d_in[4];
  const float* mass_w   = (const float*)d_in[5];
  const float* encWqkv  = (const float*)d_in[6];
  const float* encbqkv  = (const float*)d_in[7];
  const float* encWo    = (const float*)d_in[8];
  const float* encbo    = (const float*)d_in[9];
  const float* encg1    = (const float*)d_in[10];
  const float* encb1    = (const float*)d_in[11];
  const float* encg2    = (const float*)d_in[12];
  const float* encb2    = (const float*)d_in[13];
  const float* encW1    = (const float*)d_in[14];
  const float* encbb1   = (const float*)d_in[15];
  const float* encW2    = (const float*)d_in[16];
  const float* encbb2   = (const float*)d_in[17];
  const float* tcaWq    = (const float*)d_in[18];
  const float* tcabq    = (const float*)d_in[19];
  const float* tcaWk    = (const float*)d_in[20];
  const float* tcabk    = (const float*)d_in[21];
  const float* tcaWv    = (const float*)d_in[22];
  const float* tcabv    = (const float*)d_in[23];
  const float* tcaWo    = (const float*)d_in[24];
  const float* tcabo    = (const float*)d_in[25];
  const float* tcag1    = (const float*)d_in[26];
  const float* tcab1    = (const float*)d_in[27];
  const float* tcaW1    = (const float*)d_in[28];
  const float* tcabb1   = (const float*)d_in[29];
  const float* tcaW2    = (const float*)d_in[30];
  const float* tcabb2   = (const float*)d_in[31];
  const float* tcag2    = (const float*)d_in[32];
  const float* tcab2    = (const float*)d_in[33];
  const float* gtWqkv   = (const float*)d_in[34];
  const float* gtbqkv   = (const float*)d_in[35];
  const float* gtWo     = (const float*)d_in[36];
  const float* gtbo     = (const float*)d_in[37];
  const float* gtg1     = (const float*)d_in[38];
  const float* gtb1     = (const float*)d_in[39];
  const float* gtg2     = (const float*)d_in[40];
  const float* gtb2     = (const float*)d_in[41];
  const float* gtW1     = (const float*)d_in[42];
  const float* gtbb1    = (const float*)d_in[43];
  const float* gtW2     = (const float*)d_in[44];
  const float* gtbb2    = (const float*)d_in[45];
  const float* ph_g     = (const float*)d_in[46];
  const float* ph_b     = (const float*)d_in[47];
  const float* sW1      = (const float*)d_in[48];
  const float* sb1      = (const float*)d_in[49];
  const float* sW2      = (const float*)d_in[50];
  const float* sb2      = (const float*)d_in[51];
  const float* sW3      = (const float*)d_in[52];
  const float* sb3      = (const float*)d_in[53];

  // -------- workspace carve-out --------
  unsigned char* base = (unsigned char*)d_ws;
  size_t off = 0;
  auto alloc = [&](size_t bytes) -> unsigned char* {
    off = (off + 255) & ~(size_t)255;
    unsigned char* p = base + off;
    off += bytes;
    return p;
  };
  const int MR  = NB * NJ;            // 14336 jet rows
  const int MT  = NB * NTRIP;         // 71680 triplet rows
  int* t_trips = (int*)alloc(NTRIP * 3 * 4);
  int* t_isr   = (int*)alloc(NASGN * 4);
  int* t_g1t   = (int*)alloc(NASGN * 4);
  int* t_g2t   = (int*)alloc(NASGN * 4);
  int* t_g1j   = (int*)alloc(NASGN * 3 * 4);
  int* t_g2j   = (int*)alloc(NASGN * 3 * 4);
  unsigned short* wEncQkv = (unsigned short*)alloc((size_t)NL * 384 * ND * 2);
  unsigned short* wEncWo  = (unsigned short*)alloc((size_t)NL * ND * ND * 2);
  unsigned short* wEncW1  = (unsigned short*)alloc((size_t)NL * NDFF * ND * 2);
  unsigned short* wEncW2  = (unsigned short*)alloc((size_t)NL * ND * NDFF * 2);
  unsigned short* wTcaQ   = (unsigned short*)alloc((size_t)ND * ND * 2);
  unsigned short* wTcaK   = (unsigned short*)alloc((size_t)ND * ND * 2);
  unsigned short* wTcaV   = (unsigned short*)alloc((size_t)ND * ND * 2);
  unsigned short* wTcaO   = (unsigned short*)alloc((size_t)ND * ND * 2);
  unsigned short* wTcaW1  = (unsigned short*)alloc((size_t)256 * ND * 2);
  unsigned short* wTcaW2  = (unsigned short*)alloc((size_t)ND * 256 * 2);
  unsigned short* wGtQkv  = (unsigned short*)alloc((size_t)384 * ND * 2);
  unsigned short* wGtWo   = (unsigned short*)alloc((size_t)ND * ND * 2);
  unsigned short* wGtW1   = (unsigned short*)alloc((size_t)256 * ND * 2);
  unsigned short* wGtW2   = (unsigned short*)alloc((size_t)ND * 256 * 2);
  unsigned short* wS1     = (unsigned short*)alloc((size_t)256 * KPAD_S1 * 2);
  unsigned short* wS2     = (unsigned short*)alloc((size_t)ND * 256 * 2);
  float*          xf      = (float*)alloc((size_t)MR * ND * 4);
  unsigned short* hb      = (unsigned short*)alloc((size_t)MR * ND * 2);
  float*          qkv     = (float*)alloc((size_t)MR * 384 * 4);
  unsigned short* attnb   = (unsigned short*)alloc((size_t)MR * ND * 2);
  unsigned short* ffh     = (unsigned short*)alloc((size_t)MR * NDFF * 2);
  float*          biasb   = (float*)alloc((size_t)NB * NH * 49 * 4);
  unsigned short* xb      = (unsigned short*)alloc((size_t)MR * ND * 2);
  float*          qbuf    = (float*)alloc((size_t)MR * ND * 4);
  unsigned short* tripb   = (unsigned short*)alloc((size_t)MT * ND * 2);
  unsigned short* kbuf    = (unsigned short*)alloc((size_t)MT * ND * 2);
  unsigned short* vbuf    = (unsigned short*)alloc((size_t)MT * ND * 2);
  float*          ybuf    = (float*)alloc((size_t)MR * ND * 4);
  unsigned short* ybb     = (unsigned short*)alloc((size_t)MR * ND * 2);
  float*          zbuf    = (float*)alloc((size_t)MR * ND * 4);
  float*          gemb    = (float*)alloc((size_t)MT * ND * 4);
  float*          evm     = (float*)alloc((size_t)NB * ND * 4);
  float*          phys    = (float*)alloc((size_t)NB * NASGN * 24 * 4);

  auto wprep = [&](const float* W, unsigned short* dst, int K, int N, int Kpad) {
    int total = N * Kpad;
    wprep_kernel<<<(total + 255) / 256, 256, 0, stream>>>(W, dst, K, N, Kpad);
  };
  auto gemm = [&](const unsigned short* A, const unsigned short* Bt,
                  const float* bias, void* out, int M, int N, int K, int mode) {
    dim3 g(N / 64, M / 16);
    switch (mode) {
      case 0: gemm_bf16_kernel<0><<<g, 32, 0, stream>>>(A, Bt, bias, out, M, N, K); break;
      case 1: gemm_bf16_kernel<1><<<g, 32, 0, stream>>>(A, Bt, bias, out, M, N, K); break;
      case 2: gemm_bf16_kernel<2><<<g, 32, 0, stream>>>(A, Bt, bias, out, M, N, K); break;
      case 3: gemm_bf16_kernel<3><<<g, 32, 0, stream>>>(A, Bt, bias, out, M, N, K); break;
      default: gemm_bf16_kernel<4><<<g, 32, 0, stream>>>(A, Bt, bias, out, M, N, K); break;
    }
  };

  // tables + weight prep
  build_tables_kernel<<<1, 32, 0, stream>>>(t_trips, t_isr, t_g1t, t_g2t, t_g1j, t_g2j);
  for (int l = 0; l < NL; ++l) {
    wprep(encWqkv + (size_t)l * ND * 384, wEncQkv + (size_t)l * 384 * ND, ND, 384, ND);
    wprep(encWo   + (size_t)l * ND * ND,  wEncWo  + (size_t)l * ND * ND,  ND, ND, ND);
    wprep(encW1   + (size_t)l * ND * NDFF, wEncW1 + (size_t)l * NDFF * ND, ND, NDFF, ND);
    wprep(encW2   + (size_t)l * NDFF * ND, wEncW2 + (size_t)l * ND * NDFF, NDFF, ND, NDFF);
  }
  wprep(tcaWq, wTcaQ, ND, ND, ND);   wprep(tcaWk, wTcaK, ND, ND, ND);
  wprep(tcaWv, wTcaV, ND, ND, ND);   wprep(tcaWo, wTcaO, ND, ND, ND);
  wprep(tcaW1, wTcaW1, ND, 256, ND); wprep(tcaW2, wTcaW2, 256, ND, 256);
  wprep(gtWqkv, wGtQkv, ND, 384, ND); wprep(gtWo, wGtWo, ND, ND, ND);
  wprep(gtW1, wGtW1, ND, 256, ND);    wprep(gtW2, wGtW2, 256, ND, 256);
  wprep(sW1, wS1, 664, 256, KPAD_S1); wprep(sW2, wS2, 256, ND, 256);

  // input embed + pairwise bias
  featurize_kernel<<<NB, 128, 0, stream>>>(fm, W_in, b_in, pt_w, dr_w, mass_w, xf, biasb);

  // encoder stack
  for (int l = 0; l < NL; ++l) {
    ln_kernel<false><<<MR, 128, 0, stream>>>(xf, encg1 + l * ND, encb1 + l * ND, nullptr, hb);
    gemm(hb, wEncQkv + (size_t)l * 384 * ND, encbqkv + l * 384, qkv, MR, 384, ND, 0);
    enc_attn_kernel<<<NB, 64, 0, stream>>>(qkv, biasb, attnb);
    gemm(attnb, wEncWo + (size_t)l * ND * ND, encbo + l * ND, xf, MR, ND, ND, 4);
    ln_kernel<false><<<MR, 128, 0, stream>>>(xf, encg2 + l * ND, encb2 + l * ND, nullptr, hb);
    gemm(hb, wEncW1 + (size_t)l * NDFF * ND, encbb1 + l * NDFF, ffh, MR, NDFF, ND, 2);
    gemm(ffh, wEncW2 + (size_t)l * ND * NDFF, encbb2 + l * ND, xf, MR, ND, NDFF, 4);
  }

  // triplet cross-attention block
  f32_to_bf16_kernel<<<(MR * ND + 255) / 256, 256, 0, stream>>>(xf, xb, MR * ND);
  gemm(xb, wTcaQ, tcabq, qbuf, MR, ND, ND, 0);
  triplet_mean_kernel<<<(MT * ND + 255) / 256, 256, 0, stream>>>(xf, t_trips, tripb);
  gemm(tripb, wTcaK, tcabk, kbuf, MT, ND, ND, 1);
  gemm(tripb, wTcaV, tcabv, vbuf, MT, ND, ND, 1);
  tca_attn_kernel<<<NB, 32, 0, stream>>>(qbuf, kbuf, vbuf, t_trips, attnb);
  gemm(attnb, wTcaO, tcabo, xf, MR, ND, ND, 4);
  ln_kernel<true><<<MR, 128, 0, stream>>>(xf, tcag1, tcab1, ybuf, ybb);
  gemm(ybb, wTcaW1, tcabb1, ffh, MR, 256, ND, 3);      // gelu
  gemm(ffh, wTcaW2, tcabb2, ybuf, MR, ND, 256, 4);
  ln_kernel<true><<<MR, 128, 0, stream>>>(ybuf, tcag2, tcab2, zbuf, hb);

  // fused group transformer over all B*35 triplet sequences
  gt_fused_kernel<<<MT / 16, 128, 0, stream>>>(
      zbuf, wGtQkv, gtbqkv, wGtWo, gtbo, wGtW1, gtbb1, wGtW2, gtbb2,
      gtg1, gtb1, gtg2, gtb2, t_trips, gemb);

  // scoring inputs
  event_mean_kernel<<<NB, 128, 0, stream>>>(zbuf, evm);
  physics_kernel<<<(NB * NASGN + 255) / 256, 256, 0, stream>>>(
      fm, t_isr, t_g1j, t_g2j, ph_g, ph_b, phys);

  // fused scoring MLP -> d_out (B, 70) f32
  score_fused_kernel<<<(NB * NASGN) / 16, 128, 0, stream>>>(
      zbuf, gemb, evm, phys, wS1, sb1, wS2, sb2, sW3, sb3,
      t_isr, t_g1t, t_g2t, (float*)d_out);
}
